// Rttention_41214506172933
// MI455X (gfx1250) — matrix operand order repaired and checked by reading, not compiled, not run on hardware
//
#include <hip/hip_runtime.h>
#include <hip/hip_bf16.h>
#include <math.h>

typedef __bf16  bf16_t;
typedef __bf16  v16bf  __attribute__((ext_vector_type(16)));
typedef __bf16  bf16x8 __attribute__((ext_vector_type(8)));
typedef __bf16  bf16x2 __attribute__((ext_vector_type(2)));
typedef float   v8f    __attribute__((ext_vector_type(8)));

#define HW 16384
#define HDIM 128

// ---------------------------------------------------------------------------
// Zero-init small fp32 scratch (sumsq accumulators)
// ---------------------------------------------------------------------------
__global__ void zero_f32(float* p, int n) {
    for (int i = threadIdx.x + blockIdx.x * blockDim.x; i < n; i += blockDim.x * gridDim.x)
        p[i] = 0.0f;
}

// ---------------------------------------------------------------------------
// FiLM modulation: mod[b][c] = 1 + (k_v[b,:192] @ w_kR.T | k_v[b,192:] @ w_kI.T)
// ---------------------------------------------------------------------------
__global__ void film_mod(const float* __restrict__ kv, const float* __restrict__ wkR,
                         const float* __restrict__ wkI, float* __restrict__ mod) {
    const int b = blockIdx.x;
    const int c = threadIdx.x;   // 0..255
    float acc = 0.0f;
    if (c < 192) {
        const float* kvb = kv + b * 256;
        const float* wr  = wkR + c * 192;
        for (int j = 0; j < 192; ++j) acc += kvb[j] * wr[j];
    } else {
        const int c2 = c - 192;
        const float* kvb = kv + b * 256 + 192;
        const float* wr  = wkI + c2 * 64;
        for (int j = 0; j < 64; ++j) acc += kvb[j] * wr[j];
    }
    mod[b * 256 + c] = 1.0f + acc;
}

// ---------------------------------------------------------------------------
// Tiled WMMA GEMM:  Y[b][m][n] = sum_k W[m][k] * (X[b][k][n] * mod[k])
// Block = 128 threads = 4 waves; tile 64(M) x 64(N), K step 32.
// LDS tiles stored in fragment-native layout so every fragment is two
// contiguous 16B runs -> ds_load_b128 instead of scalar u16 gathers.
//   As [m][k]   row-major, 40-elem (80B) row stride
//   BsT[n][k]   TRANSPOSED (k contiguous), 40-elem row stride
// ---------------------------------------------------------------------------
template <bool XBF16, bool USEMOD, bool OUTF32>
__global__ __launch_bounds__(128) void conv1x1_wmma(
    const float* __restrict__ Wt,     // [Mtot x Ktot] fp32 row-major
    const void*  __restrict__ Xv,     // [Ktot x N] fp32 or bf16 (after offsets)
    const float* __restrict__ mod,    // [4*256] FiLM scales (or nullptr)
    int modOff,
    void* __restrict__ Yv,            // [Mtot x N] bf16 or fp32
    int Mtot, int Ktot, int N,
    long long xRowOff, long long xBatchStride, long long yBatchStride)
{
    const int tid  = threadIdx.x;
    const int lane = tid & 31;
    const int wave = tid >> 5;
    const int b    = blockIdx.z;
    const int m0   = blockIdx.y * 64;
    const int n0   = blockIdx.x * 64;

    const float*  Xf = (const float*)Xv  + b * xBatchStride + xRowOff;
    const bf16_t* Xb = (const bf16_t*)Xv + b * xBatchStride + xRowOff;

    __shared__ alignas(16) bf16_t As [64][40];   // [m][k]
    __shared__ alignas(16) bf16_t BsT[64][40];   // [n][k] (transposed)

    v8f acc[4] = {};

    // A staging map: thread -> (row 0..63, 16-col half)
    const int arow  = tid >> 1;
    const int acol0 = (tid & 1) * 16;
    // B staging map: thread -> (k-pair 0..15, n-octet 0..7)
    const int kp  = tid & 15;        // k = 2*kp, 2*kp+1
    const int ng  = tid >> 4;        // n block of 8
    const int nb8 = ng * 8;

    for (int k0 = 0; k0 < Ktot; k0 += 32) {
        // ---- stage A (weights, fp32 -> bf16), two ds_store_b128 per thread ----
        {
            const float* wrow = Wt + (long long)(m0 + arow) * Ktot + k0 + acol0;
            bf16x8 v0, v1;
            #pragma unroll
            for (int j = 0; j < 8; ++j) { v0[j] = (bf16_t)wrow[j]; v1[j] = (bf16_t)wrow[8 + j]; }
            *(bf16x8*)&As[arow][acol0]     = v0;
            *(bf16x8*)&As[arow][acol0 + 8] = v1;
        }

        // ---- stage B transposed: pack (k,k+1) pairs, ds_store_b32 x8 ----
        {
            const long long xoff = (long long)(k0 + 2 * kp) * N + n0 + nb8;
            float m0v = 1.0f, m1v = 1.0f;
            if (USEMOD) {
                m0v = mod[b * 256 + modOff + k0 + 2 * kp];
                m1v = mod[b * 256 + modOff + k0 + 2 * kp + 1];
            }
            #pragma unroll
            for (int j = 0; j < 8; ++j) {
                float x0 = XBF16 ? (float)Xb[xoff + j]     : Xf[xoff + j];
                float x1 = XBF16 ? (float)Xb[xoff + N + j] : Xf[xoff + N + j];
                bf16x2 p;
                p[0] = (bf16_t)(USEMOD ? x0 * m0v : x0);
                p[1] = (bf16_t)(USEMOD ? x1 * m1v : x1);
                *(bf16x2*)&BsT[nb8 + j][2 * kp] = p;
            }
            if (k0 + 32 < Ktot) {   // prefetch next K panel (global_prefetch_b8)
                if (XBF16) __builtin_prefetch(Xb + xoff + 32LL * N, 0, 1);
                else       __builtin_prefetch(Xf + xoff + 32LL * N, 0, 1);
            }
        }
        __syncthreads();

        // ---- A fragment: lanes 0-15 K {0..7,16..23}, lanes 16-31 K {8..15,24..31}
        const int mrow = 16 * wave + (lane & 15);
        const int ka   = (lane < 16) ? 0 : 8;
        const bf16x8 a0 = *(const bf16x8*)&As[mrow][ka];        // ds_load_b128
        const bf16x8 a1 = *(const bf16x8*)&As[mrow][ka + 16];   // ds_load_b128
        v16bf a;
        #pragma unroll
        for (int i = 0; i < 8; ++i) { a[i] = a0[i]; a[8 + i] = a1[i]; }

        // ---- 4 B fragments (contiguous in BsT) + WMMA ----
        const int kb = (lane < 16) ? 0 : 8;
        #pragma unroll
        for (int s = 0; s < 4; ++s) {
            const int nn = 16 * s + (lane & 15);
            const bf16x8 b0 = *(const bf16x8*)&BsT[nn][kb];      // ds_load_b128
            const bf16x8 b1 = *(const bf16x8*)&BsT[nn][kb + 16]; // ds_load_b128 (K +16)
            v16bf bb;
            #pragma unroll
            for (int i = 0; i < 8; ++i) { bb[i] = b0[i]; bb[8 + i] = b1[i]; }
            acc[s] = __builtin_amdgcn_wmma_f32_16x16x32_bf16(
                false, a, false, bb, (short)0, acc[s], false, false);
        }
        __syncthreads();
    }

    // ---- epilogue: C layout lane<16 -> M rows 0..7, lane>=16 -> M rows 8..15
    const int rowBase = m0 + 16 * wave + ((lane < 16) ? 0 : 8);
    const int nbi     = lane & 15;
    if (OUTF32) {
        float* Y = (float*)Yv + b * yBatchStride;
        #pragma unroll
        for (int s = 0; s < 4; ++s)
            #pragma unroll
            for (int j = 0; j < 8; ++j)
                Y[(long long)(rowBase + j) * N + n0 + 16 * s + nbi] = acc[s][j];
    } else {
        bf16_t* Y = (bf16_t*)Yv + b * yBatchStride;
        #pragma unroll
        for (int s = 0; s < 4; ++s)
            #pragma unroll
            for (int j = 0; j < 8; ++j)
                Y[(long long)(rowBase + j) * N + n0 + 16 * s + nbi] = (bf16_t)acc[s][j];
    }
}

// ---------------------------------------------------------------------------
// Depthwise 3x3 (SAME, zero pad) + optional per-channel sum-of-squares
// ---------------------------------------------------------------------------
__global__ __launch_bounds__(256) void dwconv3x3_k(
    const bf16_t* __restrict__ in, const float* __restrict__ w9,
    bf16_t* __restrict__ out, float* __restrict__ sumsq,
    int C, int sumsqLimit)
{
    const int tid = threadIdx.x;
    const int b = blockIdx.z, c = blockIdx.y;
    const int p = blockIdx.x * 256 + tid;
    const int y = p >> 7, x = p & 127;
    const long long base = ((long long)b * C + c) * HW;
    const float* wr = w9 + c * 9;

    float acc = 0.0f;
    #pragma unroll
    for (int dy = -1; dy <= 1; ++dy)
        #pragma unroll
        for (int dx = -1; dx <= 1; ++dx) {
            const int yy = y + dy, xx = x + dx;
            if (yy >= 0 && yy < HDIM && xx >= 0 && xx < HDIM)
                acc += wr[(dy + 1) * 3 + (dx + 1)] * (float)in[base + yy * HDIM + xx];
        }
    out[base + p] = (bf16_t)acc;

    if (sumsq != nullptr && c < sumsqLimit) {
        __shared__ float red[256];
        red[tid] = acc * acc;
        __syncthreads();
        for (int s = 128; s > 0; s >>= 1) {
            if (tid < s) red[tid] += red[tid + s];
            __syncthreads();
        }
        if (tid == 0) atomicAdd(&sumsq[b * sumsqLimit + c], red[0]);
    }
}

// ---------------------------------------------------------------------------
// Channel attention logits: raw[b][h][c][d] = sum_n q[c,n]*k[d,n]
// Vectorized bf16x8 (b128) global loads; one block per (c,h,b).
// ---------------------------------------------------------------------------
__global__ __launch_bounds__(256) void attn_dot_k(
    const bf16_t* __restrict__ q, const bf16_t* __restrict__ kvo,
    float* __restrict__ raw)
{
    const int c = blockIdx.x, h = blockIdx.y, b = blockIdx.z, tid = threadIdx.x;
    const bf16_t* qr = q   + ((long long)(b * 256 + h * 32 + c)) * HW;
    const bf16_t* kb = kvo + ((long long)(b * 512 + h * 32)) * HW;  // k = first 256 ch

    float acc[32];
    #pragma unroll
    for (int d = 0; d < 32; ++d) acc[d] = 0.0f;

    for (int nb = tid * 8; nb < HW; nb += 256 * 8) {
        const bf16x8 q8 = *(const bf16x8*)&qr[nb];
        float qf[8];
        #pragma unroll
        for (int j = 0; j < 8; ++j) qf[j] = (float)q8[j];
        #pragma unroll
        for (int d = 0; d < 32; ++d) {
            const bf16x8 k8 = *(const bf16x8*)&kb[(long long)d * HW + nb];
            float s = 0.0f;
            #pragma unroll
            for (int j = 0; j < 8; ++j) s += qf[j] * (float)k8[j];
            acc[d] += s;
        }
    }

    __shared__ float red[256];
    for (int d = 0; d < 32; ++d) {
        red[tid] = acc[d];
        __syncthreads();
        for (int s = 128; s > 0; s >>= 1) {
            if (tid < s) red[tid] += red[tid + s];
            __syncthreads();
        }
        if (tid == 0) raw[(((long long)(b * 8 + h) * 32) + c) * 32 + d] = red[0];
        __syncthreads();
    }
}

// ---------------------------------------------------------------------------
// Scale by 1/||q||, 1/||k||, temperature; softmax over d (32 wide)
// ---------------------------------------------------------------------------
__global__ __launch_bounds__(32) void softmax_scale_k(
    const float* __restrict__ raw, const float* __restrict__ sqq,
    const float* __restrict__ sqk, const float* __restrict__ temp,
    float* __restrict__ attn)
{
    const int bh = blockIdx.x;         // b*8+h
    const int b = bh >> 3, h = bh & 7;
    const int c = threadIdx.x;         // 0..31
    const float invq = 1.0f / fmaxf(sqrtf(sqq[b * 256 + h * 32 + c]), 1e-12f);
    const float t = temp[h];
    const float* r = raw + ((long long)bh * 32 + c) * 32;

    float av[32], mx = -3.4e38f;
    #pragma unroll
    for (int d = 0; d < 32; ++d) {
        const float invk = 1.0f / fmaxf(sqrtf(sqk[b * 256 + h * 32 + d]), 1e-12f);
        const float v = r[d] * invq * invk * t;
        av[d] = v;
        mx = fmaxf(mx, v);
    }
    float s = 0.0f;
    #pragma unroll
    for (int d = 0; d < 32; ++d) { av[d] = __expf(av[d] - mx); s += av[d]; }
    const float inv = 1.0f / s;
    float* o = attn + ((long long)bh * 32 + c) * 32;
    #pragma unroll
    for (int d = 0; d < 32; ++d) o[d] = av[d] * inv;
}

// ---------------------------------------------------------------------------
// out_attn[b][h][c][n] = sum_d attn[c][d] * v[d][n]   (v = channels 256.. of kv)
// ---------------------------------------------------------------------------
__global__ __launch_bounds__(256) void attn_v_k(
    const float* __restrict__ attn, const bf16_t* __restrict__ kvo,
    bf16_t* __restrict__ outa)
{
    const int tid = threadIdx.x;
    const int h = blockIdx.y, b = blockIdx.z;
    __shared__ float Am[1024];
    const float* ab = attn + ((long long)(b * 8 + h)) * 1024;
    for (int i = tid; i < 1024; i += 256) Am[i] = ab[i];
    __syncthreads();

    const long long n = (long long)blockIdx.x * 256 + tid;
    const bf16_t* vb = kvo + ((long long)(b * 512 + 256 + h * 32)) * HW;
    float vr[32];
    #pragma unroll
    for (int d = 0; d < 32; ++d) vr[d] = (float)vb[(long long)d * HW + n];

    bf16_t* ob = outa + ((long long)(b * 256 + h * 32)) * HW;
    for (int c = 0; c < 32; ++c) {
        float acc = 0.0f;
        #pragma unroll
        for (int d = 0; d < 32; ++d) acc += Am[c * 32 + d] * vr[d];
        ob[(long long)c * HW + n] = (bf16_t)acc;
    }
}

// ---------------------------------------------------------------------------
// Host-side launch
// ---------------------------------------------------------------------------
extern "C" void kernel_launch(void* const* d_in, const int* in_sizes, int n_in,
                              void* d_out, int out_size, void* d_ws, size_t ws_size,
                              hipStream_t stream) {
    (void)in_sizes; (void)n_in; (void)out_size; (void)ws_size;
    const float* x       = (const float*)d_in[0];  // [4,256,128,128]
    const float* k_v     = (const float*)d_in[1];  // [4,256]
    const float* temp    = (const float*)d_in[2];  // [8,1,1]
    const float* w_kR    = (const float*)d_in[3];  // [192,192]
    const float* w_kI    = (const float*)d_in[4];  // [64,64]
    const float* w_qR    = (const float*)d_in[5];  // [256,192]
    const float* w_qdw   = (const float*)d_in[6];  // [256,1,3,3]
    const float* w_kvI   = (const float*)d_in[7];  // [512,64]
    const float* w_kvdw  = (const float*)d_in[8];  // [512,1,3,3]
    const float* w_proj  = (const float*)d_in[9];  // [256,256]
    float* out = (float*)d_out;                    // [4,256,128,128]

    char* ws = (char*)d_ws;
    bf16_t* q_pre    = (bf16_t*)(ws + 0);           // 32MB
    bf16_t* kv_pre   = (bf16_t*)(ws + 33554432);    // 64MB
    bf16_t* qbuf     = (bf16_t*)(ws + 100663296);   // 32MB
    bf16_t* kvout    = (bf16_t*)(ws + 134217728);   // 64MB (k: ch 0..255, v: 256..511)
    bf16_t* out_attn = (bf16_t*)(ws + 0);           // reuse q_pre space
    char* small = ws + 201326592;
    float* modbuf   = (float*)(small);              // [4*256]
    float* sq_q     = (float*)(small + 4096);       // [4*256]
    float* sq_k     = (float*)(small + 8192);       // [4*256]
    float* attn_raw = (float*)(small + 12288);      // [4*8*32*32]
    float* attn     = (float*)(small + 12288 + 131072);

    const long long HWL = HW;

    zero_f32<<<4, 256, 0, stream>>>(sq_q, 2048);
    film_mod<<<4, 256, 0, stream>>>(k_v, w_kR, w_kI, modbuf);

    conv1x1_wmma<false, true, false><<<dim3(256, 4, 4), 128, 0, stream>>>(
        w_qR, x, modbuf, 0, q_pre, 256, 192, HW,
        0LL, 256LL * HWL, 256LL * HWL);

    conv1x1_wmma<false, true, false><<<dim3(256, 8, 4), 128, 0, stream>>>(
        w_kvI, x, modbuf, 192, kv_pre, 512, 64, HW,
        192LL * HWL, 256LL * HWL, 512LL * HWL);

    dwconv3x3_k<<<dim3(64, 256, 4), 256, 0, stream>>>(q_pre, w_qdw, qbuf, sq_q, 256, 256);
    dwconv3x3_k<<<dim3(64, 512, 4), 256, 0, stream>>>(kv_pre, w_kvdw, kvout, sq_k, 512, 256);

    attn_dot_k<<<dim3(32, 8, 4), 256, 0, stream>>>(qbuf, kvout, attn_raw);
    softmax_scale_k<<<32, 32, 0, stream>>>(attn_raw, sq_q, sq_k, temp, attn);
    attn_v_k<<<dim3(64, 8, 4), 256, 0, stream>>>(attn, kvout, out_attn);

    conv1x1_wmma<true, false, true><<<dim3(256, 4, 4), 128, 0, stream>>>(
        w_proj, out_attn, nullptr, 0, out, 256, 256, HW,
        0LL, 256LL * HWL, 256LL * HWL);
}
